// Attention_79688823210300
// MI455X (gfx1250) — matrix operand order repaired and checked by reading, not compiled, not run on hardware
//
#include <hip/hip_runtime.h>
#include <hip/hip_bf16.h>
#include <math.h>

// ---------- types ----------
typedef __attribute__((ext_vector_type(16))) __bf16      v16bf;
typedef __attribute__((ext_vector_type(8)))  float       v8f;
typedef __attribute__((ext_vector_type(4)))  unsigned int u32x4;

union FragBF { v16bf v; u32x4 q[2]; };

#define NB   128
#define NSEQ 243
#define NC   512
#define NH   8
#define NHD  64
#define MTOT (NB * NSEQ)          // 31104
#define OUTSZ ((size_t)MTOT * NC) // 15925248 elements per output tensor

// ---------- helpers ----------
__device__ __forceinline__ unsigned short f2bf(float f) {
    unsigned int u = __float_as_uint(f);
    u += 0x7FFFu + ((u >> 16) & 1u);          // round-to-nearest-even
    return (unsigned short)(u >> 16);
}
__device__ __forceinline__ float gelu_exact(float x) {
    return 0.5f * x * (1.0f + erff(x * 0.70710678118654752f));
}
__device__ __forceinline__ float wave_sum32(float v) {
#pragma unroll
    for (int m = 16; m >= 1; m >>= 1) v += __shfl_xor(v, m, 32);
    return v;
}
__device__ __forceinline__ float half_max16(float v) {
#pragma unroll
    for (int m = 8; m >= 1; m >>= 1) v = fmaxf(v, __shfl_xor(v, m, 32));
    return v;
}
__device__ __forceinline__ float half_sum16(float v) {
#pragma unroll
    for (int m = 8; m >= 1; m >>= 1) v += __shfl_xor(v, m, 32);
    return v;
}
__device__ __forceinline__ v16bf load2x16B(const unsigned short* p0,
                                           const unsigned short* p1) {
    FragBF f;
    f.q[0] = *(const u32x4*)p0;
    f.q[1] = *(const u32x4*)p1;
    return f.v;
}
__device__ __forceinline__ v8f wmma_bf16(v16bf a, v16bf b, v8f c) {
    return __builtin_amdgcn_wmma_f32_16x16x32_bf16(false, a, false, b,
                                                   (short)0, c, false, false);
}
// CDNA5 async memory->LDS copy (GV mode), tracked by ASYNCcnt.
// lds_off = byte offset within LDS (low 32 bits of flat shared pointer),
// gaddr   = 64-bit global address.
__device__ __forceinline__ void async_ld_lds_b128(unsigned lds_off,
                                                  unsigned long long gaddr) {
    asm volatile("global_load_async_to_lds_b128 %0, %1, off"
                 :: "v"(lds_off), "v"(gaddr) : "memory");
}
__device__ __forceinline__ void wait_asynccnt0() {
    asm volatile("s_wait_asynccnt 0" ::: "memory");
}
__device__ __forceinline__ unsigned lds_off32(const void* p) {
    return (unsigned)(unsigned long long)p;   // flat shared ptr: low word = LDS offset
}

// ---------- f32 -> bf16 convert ----------
__global__ __launch_bounds__(256)
void cvt_f32_bf16(const float* __restrict__ src, unsigned short* __restrict__ dst, int n) {
    int i = blockIdx.x * 256 + threadIdx.x;
    if (i < n) dst[i] = f2bf(src[i]);
}

// ---------- pooling / conv-transpose pyramid : one wave per (b,c) row ----------
__device__ __forceinline__ void ln_gelu_inplace(float* buf, int L, float lsum, float lss,
                                                const float* g, const float* be, int lane) {
    float S  = wave_sum32(lsum);
    float SS = wave_sum32(lss);
    float mean = S / (float)L;
    float var  = SS / (float)L - mean * mean;
    float rstd = rsqrtf(var + 1e-5f);
    for (int i = lane; i < L; i += 32)
        buf[i] = gelu_exact((buf[i] - mean) * rstd * g[i] + be[i]);
}

__global__ __launch_bounds__(256)
void pool_chain_kernel(const float* __restrict__ x, float* __restrict__ xr,
                       const float* g1, const float* b1, const float* g2, const float* b2,
                       const float* g3, const float* b3,
                       const float* ug1, const float* ub1, const float* ug2, const float* ub2,
                       const float* ug3, const float* ub3,
                       const float* wu1, const float* bu1, const float* wu2, const float* bu2,
                       const float* wu3, const float* bu3) {
    __shared__ float sb[8][896];
    const int wv = threadIdx.x >> 5, lane = threadIdx.x & 31;
    const int rc = blockIdx.x * 8 + wv;      // over B*C = 65536
    const int b  = rc >> 9;                  // /512
    const int c  = rc & 511;

    float* X  = sb[wv];        // 243 (alloc 244)
    float* A1 = X  + 244;      // 121 (124)
    float* A2 = A1 + 124;      // 60  (64)
    float* A3 = A2 + 64;       // 30  (32)
    float* T1 = A3 + 32;       // 60  (64)
    float* T2 = T1 + 64;       // 121 (124)
    float* T3 = T2 + 124;      // 243 (244)

    for (int i = lane; i < 243; i += 32)
        X[i] = x[((size_t)b * 243 + i) * 512 + c];

    // ---- pool 243 -> 121, LN(g1,b1), gelu ----
    {
        float ls = 0.f, lss = 0.f;
        for (int i = lane; i < 121; i += 32) {
            int s = (i * 243) / 121;
            int e = ((i + 1) * 243 + 120) / 121;
            float a = 0.f;
            for (int t = s; t < e; ++t) a += X[t];
            a /= (float)(e - s);
            A1[i] = a; ls += a; lss += a * a;
        }
        ln_gelu_inplace(A1, 121, ls, lss, g1, b1, lane);
    }
    // ---- pool 121 -> 60 ----
    {
        float ls = 0.f, lss = 0.f;
        for (int i = lane; i < 60; i += 32) {
            int s = (i * 121) / 60;
            int e = ((i + 1) * 121 + 59) / 60;
            float a = 0.f;
            for (int t = s; t < e; ++t) a += A1[t];
            a /= (float)(e - s);
            A2[i] = a; ls += a; lss += a * a;
        }
        ln_gelu_inplace(A2, 60, ls, lss, g2, b2, lane);
    }
    // ---- pool 60 -> 30 ----
    {
        float ls = 0.f, lss = 0.f;
        for (int i = lane; i < 30; i += 32) {
            int s = (i * 60) / 30;
            int e = ((i + 1) * 60 + 29) / 30;
            float a = 0.f;
            for (int t = s; t < e; ++t) a += A2[t];
            a /= (float)(e - s);
            A3[i] = a; ls += a; lss += a * a;
        }
        ln_gelu_inplace(A3, 30, ls, lss, g3, b3, lane);
    }
    // ---- convT1: k=2, stride 2, 30 -> 60 ----
    {
        float w0 = wu1[c * 2 + 0], w1 = wu1[c * 2 + 1], bb = bu1[c];
        float ls = 0.f, lss = 0.f;
        for (int j = lane; j < 60; j += 32) {
            float a = A3[j >> 1] * ((j & 1) ? w1 : w0) + bb;
            T1[j] = a; ls += a; lss += a * a;
        }
        ln_gelu_inplace(T1, 60, ls, lss, ug1, ub1, lane);
    }
    // ---- convT2: k=3, stride 2, input (c1 + p2)[60] -> 121 ----
    {
        float w0 = wu2[c * 3 + 0], w1 = wu2[c * 3 + 1], w2 = wu2[c * 3 + 2], bb = bu2[c];
        float ls = 0.f, lss = 0.f;
        for (int j = lane; j < 121; j += 32) {
            float a = bb;
            if (j & 1) {
                int m = (j - 1) >> 1;
                a += (T1[m] + A2[m]) * w1;
            } else {
                int m = j >> 1;
                if (m < 60) a += (T1[m] + A2[m]) * w0;
                if (j >= 2) { int m2 = (j - 2) >> 1; a += (T1[m2] + A2[m2]) * w2; }
            }
            T2[j] = a; ls += a; lss += a * a;
        }
        ln_gelu_inplace(T2, 121, ls, lss, ug2, ub2, lane);
    }
    // ---- convT3: k=3, stride 2, input (c2 + p1)[121] -> 243 ----
    {
        float w0 = wu3[c * 3 + 0], w1 = wu3[c * 3 + 1], w2 = wu3[c * 3 + 2], bb = bu3[c];
        float ls = 0.f, lss = 0.f;
        for (int j = lane; j < 243; j += 32) {
            float a = bb;
            if (j & 1) {
                int m = (j - 1) >> 1;
                a += (T2[m] + A1[m]) * w1;
            } else {
                int m = j >> 1;
                if (m < 121) a += (T2[m] + A1[m]) * w0;
                if (j >= 2) { int m2 = (j - 2) >> 1; a += (T2[m2] + A1[m2]) * w2; }
            }
            T3[j] = a; ls += a; lss += a * a;
        }
        ln_gelu_inplace(T3, 243, ls, lss, ug3, ub3, lane);
    }
    // ---- x_red = c3 + x ----
    for (int i = lane; i < 243; i += 32)
        xr[((size_t)b * 243 + i) * 512 + c] = T3[i] + X[i];
}

// ---------- LayerNorm over C=512, one wave per row, outputs bf16 ----------
__global__ __launch_bounds__(256)
void ln_c_kernel(const float* __restrict__ xr, unsigned short* __restrict__ kvn,
                 const float* __restrict__ g4, const float* __restrict__ b4) {
    const int wv = threadIdx.x >> 5, lane = threadIdx.x & 31;
    const int row = blockIdx.x * 8 + wv;     // < 31104
    const float* p = xr + (size_t)row * 512;
    float v[16];
    float s = 0.f, ss = 0.f;
#pragma unroll
    for (int i = 0; i < 16; ++i) {
        float t = p[i * 32 + lane];
        v[i] = t; s += t; ss += t * t;
    }
    s  = wave_sum32(s);
    ss = wave_sum32(ss);
    float mean = s * (1.f / 512.f);
    float var  = ss * (1.f / 512.f) - mean * mean;
    float rstd = rsqrtf(var + 1e-5f);
    unsigned short* o = kvn + (size_t)row * 512;
#pragma unroll
    for (int i = 0; i < 16; ++i) {
        int idx = i * 32 + lane;
        o[idx] = f2bf((v[i] - mean) * rstd * g4[idx] + b4[idx]);
    }
}

// ---------- bf16 WMMA GEMM : C[m,n] = sum_k A[m,k]*Bw[n,k] ----------
// Double-buffered LDS, async global->LDS staging, one barrier per K-step.
// mode 0: store bf16 to q_bf with [b,h,n,d] permute
// mode 1: store f32 k/v to d_out (+bf16 copies) with permute
// mode 2: store f32 out = acc + bias
__global__ __launch_bounds__(256)
void gemm_bf16_kernel(const unsigned short* __restrict__ A,
                      const unsigned short* __restrict__ Bw,
                      int mode,
                      float* __restrict__ out_f32,
                      unsigned short* __restrict__ out_bf_a,
                      unsigned short* __restrict__ out_bf_b,
                      const float* __restrict__ bias) {
    __shared__ unsigned short As[2][128 * 32];
    __shared__ unsigned short Bs[2][128 * 32];
    const int tid  = threadIdx.x;
    const int lane = tid & 31, wv = tid >> 5;
    const int wm = wv & 3, wn = wv >> 2;       // 4 (M) x 2 (N) wave grid
    const int m0 = blockIdx.x * 128;           // 243 M-blocks
    const int n0 = blockIdx.y * 128;
    const int hl = lane >> 4, lm = lane & 15;

    // staging coordinates: each thread moves one 16-byte chunk of A and B
    const int sr = tid >> 1, sk = (tid & 1) * 16;
    const unsigned long long gA0 =
        (unsigned long long)(A  + (size_t)(m0 + sr) * 512 + sk);
    const unsigned long long gB0 =
        (unsigned long long)(Bw + (size_t)(n0 + sr) * 512 + sk);
    const unsigned ldsA[2] = { lds_off32(&As[0][sr * 32 + sk]),
                               lds_off32(&As[1][sr * 32 + sk]) };
    const unsigned ldsB[2] = { lds_off32(&Bs[0][sr * 32 + sk]),
                               lds_off32(&Bs[1][sr * 32 + sk]) };

    v8f zero = {0.f, 0.f, 0.f, 0.f, 0.f, 0.f, 0.f, 0.f};
    v8f acc[2][4];
#pragma unroll
    for (int i = 0; i < 2; ++i)
#pragma unroll
        for (int j = 0; j < 4; ++j) acc[i][j] = zero;

    // prologue: async-stage K-step 0 into buffer 0
    async_ld_lds_b128(ldsA[0], gA0);
    async_ld_lds_b128(ldsB[0], gB0);
    wait_asynccnt0();
    __syncthreads();

    int cur = 0;
    for (int k0 = 0; k0 < 512; k0 += 32) {
        const int nxt = cur ^ 1;
        if (k0 + 32 < 512) {                        // prefetch next K-step
            async_ld_lds_b128(ldsA[nxt], gA0 + (unsigned long long)(k0 + 32) * 2);
            async_ld_lds_b128(ldsB[nxt], gB0 + (unsigned long long)(k0 + 32) * 2);
        }
        const unsigned short* Ab = As[cur];
        const unsigned short* Bb = Bs[cur];
        v16bf afr[2], bfr[4];
#pragma unroll
        for (int i = 0; i < 2; ++i) {
            const unsigned short* base = Ab + (wm * 32 + i * 16 + lm) * 32;
            afr[i] = load2x16B(base + 8 * hl, base + 16 + 8 * hl);
        }
#pragma unroll
        for (int j = 0; j < 4; ++j) {
            const unsigned short* base = Bb + (wn * 64 + j * 16 + lm) * 32;
            bfr[j] = load2x16B(base + 8 * hl, base + 16 + 8 * hl);
        }
#pragma unroll
        for (int i = 0; i < 2; ++i)
#pragma unroll
            for (int j = 0; j < 4; ++j)
                acc[i][j] = wmma_bf16(afr[i], bfr[j], acc[i][j]);
        wait_asynccnt0();           // our prefetch landed
        __syncthreads();            // everyone done reading 'cur' & writing 'nxt'
        cur = nxt;
    }

    // epilogue
#pragma unroll
    for (int i = 0; i < 2; ++i)
#pragma unroll
        for (int j = 0; j < 4; ++j)
#pragma unroll
            for (int r = 0; r < 8; ++r) {
                int gm = m0 + wm * 32 + i * 16 + r + 8 * hl;
                int gn = n0 + wn * 64 + j * 16 + lm;
                float vv = acc[i][j][r];
                if (mode == 2) {
                    out_f32[(size_t)gm * 512 + gn] = vv + bias[gn];
                } else {
                    int b = gm / 243, n = gm % 243;
                    if (mode == 0) {
                        int h = gn >> 6, d = gn & 63;
                        out_bf_a[(((size_t)(b * 8 + h)) * 243 + n) * 64 + d] = f2bf(vv);
                    } else {
                        int half = gn >> 9;        // 0 = k, 1 = v
                        int o = gn & 511;
                        int h = o >> 6, d = o & 63;
                        size_t idx = (((size_t)(b * 8 + h)) * 243 + n) * 64 + d;
                        if (half == 0) { out_f32[idx] = vv;          out_bf_a[idx] = f2bf(vv); }
                        else           { out_f32[OUTSZ + idx] = vv;  out_bf_b[idx] = f2bf(vv); }
                    }
                }
            }
}

// ---------- attention : one block per (b,h), 8 waves, WMMA S and PV ----------
__global__ __launch_bounds__(256)
void attn_kernel(const unsigned short* __restrict__ qbf,
                 const unsigned short* __restrict__ kbf,
                 const unsigned short* __restrict__ vbf,
                 unsigned short* __restrict__ obf) {
    extern __shared__ unsigned short smem[];
    unsigned short* Ks = smem;                   // [256][64]
    unsigned short* Vt = smem + 256 * 64;        // [64][256] (transposed V)
    unsigned short* Ps = smem + 2 * 256 * 64;    // 8 waves * [16][256]

    const int bh = blockIdx.x;
    const int b = bh >> 3, h = bh & 7;
    const int tid = threadIdx.x, lane = tid & 31, wv = tid >> 5;
    const int hl = lane >> 4, lm = lane & 15;

    const unsigned short* kg = kbf + (size_t)bh * 243 * 64;
    const unsigned short* vg = vbf + (size_t)bh * 243 * 64;
    const unsigned short* qg = qbf + (size_t)bh * 243 * 64;

    for (int idx = tid; idx < 256 * 64; idx += 256) {
        int r = idx >> 6, d = idx & 63;
        unsigned short kv = 0, vvv = 0;
        if (r < 243) { kv = kg[r * 64 + d]; vvv = vg[r * 64 + d]; }
        Ks[idx] = kv;
        Vt[d * 256 + r] = vvv;
    }
    __syncthreads();

    unsigned short* P = Ps + wv * (16 * 256);
    v8f zero = {0.f, 0.f, 0.f, 0.f, 0.f, 0.f, 0.f, 0.f};

    for (int pass = 0; pass < 2; ++pass) {
        const int mt = wv + pass * 8;            // 16 M-tiles total
        const int r0 = mt * 16;

        // ---- S = Q K^T ----
        v8f s[16];
#pragma unroll
        for (int nt = 0; nt < 16; ++nt) s[nt] = zero;
#pragma unroll
        for (int kt = 0; kt < 2; ++kt) {
            int qr = r0 + lm; if (qr > 242) qr = 242;
            const unsigned short* qb = qg + (size_t)qr * 64 + kt * 32 + 8 * hl;
            v16bf a = load2x16B(qb, qb + 16);
#pragma unroll
            for (int nt = 0; nt < 16; ++nt) {
                const unsigned short* kb = Ks + (nt * 16 + lm) * 64 + kt * 32 + 8 * hl;
                v16bf bb = load2x16B(kb, kb + 16);
                s[nt] = wmma_bf16(a, bb, s[nt]);
            }
        }
        // ---- scale + mask ----
#pragma unroll
        for (int nt = 0; nt < 16; ++nt) {
            int col = nt * 16 + lm;
            float msk = (col >= 243) ? -1e30f : 0.f;
#pragma unroll
            for (int r = 0; r < 8; ++r) {
                float vv = s[nt][r] * 0.125f;    // SCALE = 64^-0.5
                s[nt][r] = (col >= 243) ? msk : vv;
            }
        }
        // ---- row softmax (rows live per-vgpr, cols across 16-lane halves) ----
#pragma unroll
        for (int r = 0; r < 8; ++r) {
            float m = -1e30f;
#pragma unroll
            for (int nt = 0; nt < 16; ++nt) m = fmaxf(m, s[nt][r]);
            m = half_max16(m);
            float sum = 0.f;
#pragma unroll
            for (int nt = 0; nt < 16; ++nt) {
                float e = __expf(s[nt][r] - m);
                s[nt][r] = e; sum += e;
            }
            sum = half_sum16(sum);
            float inv = 1.0f / sum;
            int prow = r + 8 * hl;
#pragma unroll
            for (int nt = 0; nt < 16; ++nt)
                P[prow * 256 + nt * 16 + lm] = f2bf(s[nt][r] * inv);
        }
        // ---- O = P V  (K = 256, N = 64) ----
        v8f oa[4];
#pragma unroll
        for (int j = 0; j < 4; ++j) oa[j] = zero;
#pragma unroll
        for (int kt2 = 0; kt2 < 8; ++kt2) {
            const unsigned short* pb = P + lm * 256 + kt2 * 32 + 8 * hl;
            v16bf a = load2x16B(pb, pb + 16);
#pragma unroll
            for (int j = 0; j < 4; ++j) {
                const unsigned short* vb = Vt + (j * 16 + lm) * 256 + kt2 * 32 + 8 * hl;
                v16bf bb = load2x16B(vb, vb + 16);
                oa[j] = wmma_bf16(a, bb, oa[j]);
            }
        }
        // ---- store o to [B*N, C] bf16 for the proj GEMM ----
#pragma unroll
        for (int j = 0; j < 4; ++j)
#pragma unroll
            for (int r = 0; r < 8; ++r) {
                int row = r0 + r + 8 * hl;
                if (row < 243) {
                    int gcol = h * 64 + j * 16 + lm;
                    obf[((size_t)(b * 243 + row)) * 512 + gcol] = f2bf(oa[j][r]);
                }
            }
    }
}

// ---------- launcher ----------
extern "C" void kernel_launch(void* const* d_in, const int* in_sizes, int n_in,
                              void* d_out, int out_size, void* d_ws, size_t ws_size,
                              hipStream_t stream) {
    const float* x     = (const float*)d_in[0];
    const float* Wq    = (const float*)d_in[1];
    const float* Wkv   = (const float*)d_in[2];
    const float* Wproj = (const float*)d_in[3];
    const float* bproj = (const float*)d_in[4];
    const float* g1 = (const float*)d_in[5],  *b1 = (const float*)d_in[6];
    const float* g2 = (const float*)d_in[7],  *b2 = (const float*)d_in[8];
    const float* g3 = (const float*)d_in[9],  *b3 = (const float*)d_in[10];
    const float* g4 = (const float*)d_in[11], *b4 = (const float*)d_in[12];
    const float* ug1 = (const float*)d_in[13], *ub1 = (const float*)d_in[14];
    const float* ug2 = (const float*)d_in[15], *ub2 = (const float*)d_in[16];
    const float* ug3 = (const float*)d_in[17], *ub3 = (const float*)d_in[18];
    const float* wu1 = (const float*)d_in[19], *bu1 = (const float*)d_in[20];
    const float* wu2 = (const float*)d_in[21], *bu2 = (const float*)d_in[22];
    const float* wu3 = (const float*)d_in[23], *bu3 = (const float*)d_in[24];

    float* out = (float*)d_out;

    // workspace layout
    char* ws = (char*)d_ws;
    size_t off = 0;
    auto alloc = [&](size_t bytes) {
        char* p = ws + off;
        off += (bytes + 255) & ~(size_t)255;
        return p;
    };
    const size_t nAct = (size_t)MTOT * NC;                  // 15,925,248
    unsigned short* x_bf    = (unsigned short*)alloc(nAct * 2);
    float*          xr      = (float*)         alloc(nAct * 4);
    unsigned short* kvn_bf  = (unsigned short*)alloc(nAct * 2);
    unsigned short* q_bf    = (unsigned short*)alloc(nAct * 2);
    unsigned short* k_bf    = (unsigned short*)alloc(nAct * 2);
    unsigned short* v_bf    = (unsigned short*)alloc(nAct * 2);
    unsigned short* o_bf    = (unsigned short*)alloc(nAct * 2);
    unsigned short* wq_bf   = (unsigned short*)alloc((size_t)512 * 512 * 2);
    unsigned short* wkv_bf  = (unsigned short*)alloc((size_t)1024 * 512 * 2);
    unsigned short* wpj_bf  = (unsigned short*)alloc((size_t)512 * 512 * 2);
    (void)ws_size; (void)n_in; (void)in_sizes; (void)out_size;

    // 1) converts
    cvt_f32_bf16<<<(int)((nAct + 255) / 256), 256, 0, stream>>>(x, x_bf, (int)nAct);
    cvt_f32_bf16<<<(512 * 512 + 255) / 256, 256, 0, stream>>>(Wq, wq_bf, 512 * 512);
    cvt_f32_bf16<<<(1024 * 512 + 255) / 256, 256, 0, stream>>>(Wkv, wkv_bf, 1024 * 512);
    cvt_f32_bf16<<<(512 * 512 + 255) / 256, 256, 0, stream>>>(Wproj, wpj_bf, 512 * 512);

    // 2) pooling / conv-transpose pyramid (B*C = 65536 rows, 8 per block)
    pool_chain_kernel<<<65536 / 8, 256, 0, stream>>>(x, xr,
        g1, b1, g2, b2, g3, b3, ug1, ub1, ug2, ub2, ug3, ub3,
        wu1, bu1, wu2, bu2, wu3, bu3);

    // 3) LN over C -> kvn (bf16)
    ln_c_kernel<<<MTOT / 8, 256, 0, stream>>>(xr, kvn_bf, g4, b4);

    // 4) Q GEMM: q = x @ Wq^T  -> q_bf [B,H,N,HD]
    gemm_bf16_kernel<<<dim3(243, 4), 256, 0, stream>>>(
        x_bf, wq_bf, 0, nullptr, q_bf, nullptr, nullptr);

    // 5) KV GEMM: kv = kvn @ Wkv^T -> k,v f32 outputs + bf16 copies
    gemm_bf16_kernel<<<dim3(243, 8), 256, 0, stream>>>(
        kvn_bf, wkv_bf, 1, out + OUTSZ, k_bf, v_bf, nullptr);

    // 6) attention -> o_bf [B*N, C]
    size_t attn_lds = (size_t)(256 * 64 + 64 * 256 + 8 * 16 * 256) * 2; // 128 KB
    attn_kernel<<<NB * NH, 256, attn_lds, stream>>>(q_bf, k_bf, v_bf, o_bf);

    // 7) proj GEMM: out = o @ Wproj^T + bproj
    gemm_bf16_kernel<<<dim3(243, 4), 256, 0, stream>>>(
        o_bf, wpj_bf, 2, out, nullptr, nullptr, bproj);
}
